// SelfAttention_85770496901684
// MI455X (gfx1250) — matrix operand order repaired and checked by reading, not compiled, not run on hardware
//
#include <hip/hip_runtime.h>

// Self-attention (causal), B=2 H=16 S=2048 D=64, fp32 in/out.
// FlashAttention-2 style, f16 WMMA + fp32 accumulation, exp2-domain online softmax.
// Row sums of P computed by WMMA against a ones matrix (same recurrence as O).

#define SEQ   2048
#define DKDIM 64
#define BM    128                // query rows per block (16 per wave, 8 waves)
#define BN    64                 // keys per iteration
#define NQT   (SEQ / BM)         // 16 query tiles
#define NWAVE 8
#define LSTR  72                 // f16 row stride for K^T / V tiles (144B, 16B-aligned)
#define PSTR  72                 // f16 row stride for per-wave P staging

typedef __attribute__((ext_vector_type(16))) _Float16 v16h;
typedef __attribute__((ext_vector_type(8)))  _Float16 v8h;
typedef __attribute__((ext_vector_type(8)))  float    v8f;
typedef __attribute__((ext_vector_type(4)))  float    v4f;

#define LOG2E 1.44269504088896340736f

// Load 16 contiguous f16 from 16B-aligned LDS as two 16B chunks.
__device__ __forceinline__ v16h load16h(const _Float16* p) {
  const v8h* q = (const v8h*)p;
  v8h lo = q[0];
  v8h hi = q[1];
  v16h r;
  #pragma unroll
  for (int i = 0; i < 8; ++i) { r[i] = lo[i]; r[i + 8] = hi[i]; }
  return r;
}

// One 64-key tile: S = Q@K^T, online softmax, O += P@V, rowsum += P@1.
// DIAG=false: branch-free fast path (tile fully below the diagonal).
// DIAG=true:  diagonal tile; skip fully-masked sub-tiles, mask the partial one.
template<bool DIAG>
__device__ __forceinline__ void tile_body(
    int kb, int q0, int qhi, int lane, int half, int l16,
    const _Float16* sKtp, const _Float16* sVp, _Float16* sPw,
    const v16h (&aq)[2], const v16h& bones,
    v8f (&acc)[4], v8f& accl, float (&mrun)[8])
{
  // ---- S tiles ----
  v8f st[4];
  #pragma unroll
  for (int j = 0; j < 4; ++j) {
    const float init = (DIAG && (kb + j * 16 > qhi)) ? -1e30f : 0.0f;
    #pragma unroll
    for (int r = 0; r < 8; ++r) st[j][r] = init;
  }
  #pragma unroll
  for (int c = 0; c < 2; ++c) {
    #pragma unroll
    for (int j = 0; j < 4; ++j) {
      if (!DIAG || (kb + j * 16 <= qhi)) {
        v16h b;
        #pragma unroll
        for (int g = 0; g < 2; ++g)
          #pragma unroll
          for (int w = 0; w < 8; ++w)
            b[g * 8 + w] = sKtp[(size_t)(c * 32 + g * 16 + half * 8 + w) * LSTR + j * 16 + l16];
        st[j] = __builtin_amdgcn_wmma_f32_16x16x32_f16(
            false, aq[c], false, b, (short)0, st[j], false, false);
      }
    }
  }
  if (DIAG) {
    #pragma unroll
    for (int j = 0; j < 4; ++j) {
      if ((kb + j * 16 <= qhi) && (kb + j * 16 + 15 > q0)) {  // partial sub-tile
        const int key = kb + j * 16 + l16;
        #pragma unroll
        for (int r = 0; r < 8; ++r) {
          const int qrow = q0 + r + 8 * half;
          if (key > qrow) st[j][r] = -1e30f;
        }
      }
    }
  }

  // ---- online softmax (exp2 domain); row max lives in one 16-lane half ----
  float mnew[8], csc[8];
  #pragma unroll
  for (int r = 0; r < 8; ++r) {
    float mx = fmaxf(fmaxf(st[0][r], st[1][r]), fmaxf(st[2][r], st[3][r]));
    mx = fmaxf(mx, __shfl_xor(mx, 1));
    mx = fmaxf(mx, __shfl_xor(mx, 2));
    mx = fmaxf(mx, __shfl_xor(mx, 4));
    mx = fmaxf(mx, __shfl_xor(mx, 8));
    mnew[r] = fmaxf(mrun[r], mx);
    csc[r]  = exp2f(mrun[r] - mnew[r]);
    mrun[r] = mnew[r];
  }

  #pragma unroll
  for (int j = 0; j < 4; ++j)
    #pragma unroll
    for (int r = 0; r < 8; ++r)
      st[j][r] = exp2f(st[j][r] - mnew[r]);

  #pragma unroll
  for (int r = 0; r < 8; ++r) {
    accl[r] *= csc[r];
    #pragma unroll
    for (int j = 0; j < 4; ++j) acc[j][r] *= csc[r];
  }

  // ---- re-lay P (D-layout f32 -> A-layout f16) through per-wave LDS ----
  #pragma unroll
  for (int j = 0; j < 4; ++j)
    #pragma unroll
    for (int r = 0; r < 8; ++r)
      sPw[(size_t)(r + 8 * half) * PSTR + j * 16 + l16] = (_Float16)st[j][r];

  // Per-wave buffer: only this wave's DS stores must complete (no barrier).
  asm volatile("s_wait_dscnt 0" ::: "memory");

  // ---- O += P @ V and rowsum += P @ 1 ; skip all-masked k-chunks ----
  #pragma unroll
  for (int c = 0; c < 2; ++c) {
    if (!DIAG || (kb + c * 32 <= qhi)) {
      v16h ap;
      const _Float16* prow = sPw + (size_t)l16 * PSTR;
      #pragma unroll
      for (int g = 0; g < 2; ++g)
        #pragma unroll
        for (int w = 0; w < 8; ++w)
          ap[g * 8 + w] = prow[c * 32 + g * 16 + half * 8 + w];
      #pragma unroll
      for (int j = 0; j < 4; ++j) {
        v16h bv;
        #pragma unroll
        for (int g = 0; g < 2; ++g)
          #pragma unroll
          for (int w = 0; w < 8; ++w)
            bv[g * 8 + w] = sVp[(size_t)(c * 32 + g * 16 + half * 8 + w) * LSTR + j * 16 + l16];
        acc[j] = __builtin_amdgcn_wmma_f32_16x16x32_f16(
            false, ap, false, bv, (short)0, acc[j], false, false);
      }
      accl = __builtin_amdgcn_wmma_f32_16x16x32_f16(
          false, ap, false, bones, (short)0, accl, false, false);
    }
  }
}

__global__ __launch_bounds__(256)
void fa2_causal_f16wmma(const float* __restrict__ Qg,
                        const float* __restrict__ Kg,
                        const float* __restrict__ Vg,
                        float* __restrict__ Og)
{
  __shared__ alignas(16) _Float16 sKt[DKDIM][LSTR];      // K^T tile: [dim][key]
  __shared__ alignas(16) _Float16 sV [BN][LSTR];         // V tile:   [key][dim]
  __shared__ alignas(16) _Float16 sP [NWAVE][16][PSTR];  // per-wave P tile

  const int tid  = threadIdx.x;
  const int lane = tid & 31;
  // Scalarize the wave id so per-wave causal guards become s_cbranch (EXEC
  // stays all-ones through every WMMA) instead of saveexec chains.
  const int wave = __builtin_amdgcn_readfirstlane(tid >> 5);
  const int half = lane >> 4;   // 0: lanes 0-15, 1: lanes 16-31
  const int l16  = lane & 15;

  const int qtile = blockIdx.x & (NQT - 1);
  const int bh    = blockIdx.x >> 4;            // 0..31 = B*H
  const float* Qb = Qg + (size_t)bh * SEQ * DKDIM;
  const float* Kb = Kg + (size_t)bh * SEQ * DKDIM;
  const float* Vb = Vg + (size_t)bh * SEQ * DKDIM;
  float*       Ob = Og + (size_t)bh * SEQ * DKDIM;

  const int q0   = qtile * BM + wave * 16;      // this wave's first query row
  const int qhi  = q0 + 15;                     // last query row of this wave

  // ---- Q fragments in WMMA A-layout.
  // Scale = (1/sqrt(64)) * log2(e) folded in, so softmax uses exp2 directly.
  v16h aq[2];
  {
    const float  qs   = 0.125f * LOG2E;
    const float* qrow = Qb + (size_t)(q0 + l16) * DKDIM;
    #pragma unroll
    for (int c = 0; c < 2; ++c)
      #pragma unroll
      for (int g = 0; g < 2; ++g)
        #pragma unroll
        for (int w = 0; w < 8; ++w)
          aq[c][g * 8 + w] = (_Float16)(qrow[c * 32 + g * 16 + half * 8 + w] * qs);
  }

  // Ones B-matrix for WMMA row sums of P.
  v16h bones;
  #pragma unroll
  for (int i = 0; i < 16; ++i) bones[i] = (_Float16)1.0f;

  v8f acc[4];      // O accumulators: 4 D-tiles covering dims 0..63
  v8f accl;        // rowsum(P) accumulator (every column = the row sum)
  float mrun[8];
  #pragma unroll
  for (int j = 0; j < 4; ++j)
    #pragma unroll
    for (int r = 0; r < 8; ++r) acc[j][r] = 0.0f;
  #pragma unroll
  for (int r = 0; r < 8; ++r) { accl[r] = 0.0f; mrun[r] = -1e30f; }

  const _Float16* sKtp = &sKt[0][0];
  const _Float16* sVp  = &sV[0][0];
  _Float16*       sPw  = &sP[wave][0][0];

  const int nkt = 2 * qtile + 2;                // key tiles covering block's causal span
  for (int t = 0; t < nkt; ++t) {
    const int kb = t * BN;

    __syncthreads();                            // previous tile's consumers done
    // ---- stage K^T and V tiles into LDS (fp32 -> f16), 256 threads ----
    {
      const int row = tid >> 2;                 // 0..63 (key within tile)
      const int c0  = (tid & 3) * 16;           // dim quarter
      const float* kr = Kb + (size_t)(kb + row) * DKDIM + c0;
      const float* vr = Vb + (size_t)(kb + row) * DKDIM + c0;
      if (t + 1 < nkt) {                        // prefetch next tile (global_prefetch_b8)
        __builtin_prefetch(kr + BN * DKDIM, 0, 1);
        __builtin_prefetch(vr + BN * DKDIM, 0, 1);
      }
      #pragma unroll
      for (int i = 0; i < 4; ++i) {
        v4f kv = *(const v4f*)(kr + i * 4);
        v4f vv = *(const v4f*)(vr + i * 4);
        const int c = c0 + i * 4;
        sKt[c + 0][row] = (_Float16)kv[0];
        sKt[c + 1][row] = (_Float16)kv[1];
        sKt[c + 2][row] = (_Float16)kv[2];
        sKt[c + 3][row] = (_Float16)kv[3];
        sV[row][c + 0] = (_Float16)vv[0];
        sV[row][c + 1] = (_Float16)vv[1];
        sV[row][c + 2] = (_Float16)vv[2];
        sV[row][c + 3] = (_Float16)vv[3];
      }
    }
    __syncthreads();

    if (kb + BN - 1 <= q0) {
      // Fast path: tile fully below the diagonal; no masking, no inner branches.
      tile_body<false>(kb, q0, qhi, lane, half, l16, sKtp, sVp, sPw,
                       aq, bones, acc, accl, mrun);
    } else if (kb <= qhi) {
      // Diagonal tile (exactly one per wave).
      tile_body<true>(kb, q0, qhi, lane, half, l16, sKtp, sVp, sPw,
                      aq, bones, acc, accl, mrun);
    }
    // else: tile entirely above this wave's rows -> nothing to do.
  }

  // ---- normalize by WMMA-computed row sums (v_rcp + mul) and write out ----
  float inv[8];
  #pragma unroll
  for (int r = 0; r < 8; ++r) inv[r] = __builtin_amdgcn_rcpf(accl[r]);
  #pragma unroll
  for (int j = 0; j < 4; ++j)
    #pragma unroll
    for (int r = 0; r < 8; ++r) {
      const int qrow = q0 + r + 8 * half;
      Ob[(size_t)qrow * DKDIM + j * 16 + l16] = acc[j][r] * inv[r];
    }
}

extern "C" void kernel_launch(void* const* d_in, const int* in_sizes, int n_in,
                              void* d_out, int out_size, void* d_ws, size_t ws_size,
                              hipStream_t stream) {
  const float* Q = (const float*)d_in[0];
  const float* K = (const float*)d_in[1];
  const float* V = (const float*)d_in[2];
  // d_in[3] is the causal mask; causality is applied analytically in-kernel.
  float* O = (float*)d_out;
  (void)in_sizes; (void)n_in; (void)out_size; (void)d_ws; (void)ws_size;

  dim3 grid(32u * NQT);   // (B*H)=32 times 16 query tiles
  dim3 block(256u);       // 8 waves of 32
  hipLaunchKernelGGL(fa2_causal_f16wmma, grid, block, 0, stream, Q, K, V, O);
}
